// Critic_77695958385085
// MI455X (gfx1250) — matrix operand order repaired and checked by reading, not compiled, not run on hardware
//
#include <hip/hip_runtime.h>

#define B_    32
#define N_    1024
#define DEG_  8
#define E_    (B_ * N_ * DEG_)      // 262144
#define NT_   (B_ * N_)             // 32768
#define EPG_  (N_ * DEG_)           // 8192 edges per graph
#define F_IN_ 64
#define HID_  128
#define HD_   4
#define HDHID_ 512
#define NR_   16
#define K1_   820
#define K2_   656
#define NEG_  0.2f

typedef __attribute__((ext_vector_type(16))) _Float16 v16h;
typedef __attribute__((ext_vector_type(8)))  _Float16 v8h;
typedef __attribute__((ext_vector_type(8)))  float    v8f;

// ---------------------------------------------------------------------------
// f32 -> f16 conversion
// ---------------------------------------------------------------------------
__global__ void cvt_f16_kernel(const float* __restrict__ src,
                               _Float16* __restrict__ dst, int n) {
  int i = blockIdx.x * blockDim.x + threadIdx.x;
  if (i < n) dst[i] = (_Float16)src[i];
}

// ---------------------------------------------------------------------------
// Pack weight [K x 512] (row-major f32) into a contiguous-k B-fragment order for
// v_wmma_f32_16x16x32_f16: per (ntile, kstep) a 32-lane x 16-half chunk.
// Lane l holds column n = ntile*16 + (l&15), K = kstep*32 + (l>>4)*16 + i.
// ---------------------------------------------------------------------------
__global__ void pack_w_kernel(const float* __restrict__ W,
                              _Float16* __restrict__ Bp, int K) {
  int idx = blockIdx.x * blockDim.x + threadIdx.x;
  int ksteps = K >> 5;
  int total = 32 * ksteps * 32 * 16;
  if (idx >= total) return;
  int i    = idx & 15;
  int lane = (idx >> 4) & 31;
  int kk   = (idx >> 9) % ksteps;
  int nt   = (idx >> 9) / ksteps;
  int n = nt * 16 + (lane & 15);
  int k = kk * 32 + ((lane >> 4) * 16) + i;
  Bp[idx] = (_Float16)W[k * HDHID_ + n];
}

// ---------------------------------------------------------------------------
// WMMA GEMM: C[M x 512] = A[M x K] * W[K x 512] + bias, output f16.
// One 16x16 output tile per wave; 8 waves per block.
// ---------------------------------------------------------------------------
__global__ void __launch_bounds__(256)
gemm_wmma_kernel(const _Float16* __restrict__ A,
                 const _Float16* __restrict__ Bp,
                 const float* __restrict__ bias,
                 _Float16* __restrict__ C, int K) {
  const int lane = threadIdx.x & 31;
  const int wave = threadIdx.x >> 5;
  const int tile = blockIdx.x * 8 + wave;
  const int mt = tile >> 5;        // 512/16 = 32 n-tiles
  const int nt = tile & 31;
  const int ksteps = K >> 5;

  const int m = mt * 16 + (lane & 15);
  const int kbase = (lane >> 4) * 16;          // A fragment K sub-base
  const _Float16* Arow = A + (size_t)m * K;
  const _Float16* Bt = Bp + (size_t)nt * ksteps * 512;

  v8f acc = {};
  for (int ks = 0; ks < K; ks += 32) {
    union { v16h v; v8h h[2]; } a, b;
    // A 16x32 f16 layout: halves 0..7 -> K = ks+kbase+0..7,
    //                     halves 8..15 -> K = ks+kbase+8..15 (contiguous 16B)
    a.h[0] = *(const v8h*)(Arow + ks + kbase);
    a.h[1] = *(const v8h*)(Arow + ks + kbase + 8);
    const _Float16* bp = Bt + (size_t)(ks >> 5) * 512 + lane * 16;
    b.h[0] = *(const v8h*)(bp);
    b.h[1] = *(const v8h*)(bp + 8);
    acc = __builtin_amdgcn_wmma_f32_16x16x32_f16(
        false, a.v, false, b.v, (short)0, acc, false, false);
  }
  const int n = nt * 16 + (lane & 15);
  const int mb = mt * 16 + (lane >> 4) * 8;
  const float bn = bias[n];
#pragma unroll
  for (int j = 0; j < 8; j++)
    C[(size_t)(mb + j) * HDHID_ + n] = (_Float16)(acc[j] + bn);
}

// ---------------------------------------------------------------------------
// CSR build: one workgroup per graph. Deterministic per-dst edge lists.
// ---------------------------------------------------------------------------
__global__ void __launch_bounds__(1024)
csr_build_kernel(const int* __restrict__ dst,
                 int* __restrict__ rowptr, int* __restrict__ eids) {
  __shared__ int sdst[EPG_];
  __shared__ int spre[N_];
  const int g = blockIdx.x;
  const int t = threadIdx.x;
  for (int i = t; i < EPG_; i += 1024)
    sdst[i] = dst[g * EPG_ + i] - g * N_;
  __syncthreads();
  int c = 0;
  for (int i = 0; i < EPG_; i++) c += (sdst[i] == t);
  spre[t] = c;
  __syncthreads();
  for (int off = 1; off < 1024; off <<= 1) {
    int add = (t >= off) ? spre[t - off] : 0;
    __syncthreads();
    spre[t] += add;
    __syncthreads();
  }
  int excl = spre[t] - c;
  rowptr[g * N_ + t] = g * EPG_ + excl;
  if (g == 0 && t == 0) rowptr[NT_] = E_;
  int pos = g * EPG_ + excl;
  for (int i = 0; i < EPG_; i++)
    if (sdst[i] == t) eids[pos++] = g * EPG_ + i;
}

// ---------------------------------------------------------------------------
// Per-(edge, head) attention logit. ef = edge_attr @ We recomputed in-kernel.
// keep==nullptr -> all edges valid; else mask = keep[src] & keep[dst].
// ---------------------------------------------------------------------------
__global__ void edge_logits_kernel(const _Float16* __restrict__ xl,
                                   const _Float16* __restrict__ xr,
                                   const float* __restrict__ edge_attr,
                                   const float* __restrict__ We,
                                   const float* __restrict__ att,
                                   const int* __restrict__ src,
                                   const int* __restrict__ dst,
                                   const int* __restrict__ keep,
                                   float* __restrict__ logits) {
  int t = blockIdx.x * blockDim.x + threadIdx.x;
  if (t >= E_ * HD_) return;
  int e = t >> 2, h = t & 3;
  int s = src[e], d = dst[e];
  if (keep && !(keep[s] && keep[d])) { logits[t] = -1e9f; return; }
  float ea[16];
#pragma unroll
  for (int k = 0; k < 16; k++) ea[k] = edge_attr[e * 16 + k];
  const int base = h * HID_;
  const _Float16* xls = xl + (size_t)s * HDHID_ + base;
  const _Float16* xrd = xr + (size_t)d * HDHID_ + base;
  float acc = 0.f;
#pragma unroll 4
  for (int c = 0; c < HID_; c++) {
    float ef = 0.f;
#pragma unroll
    for (int k = 0; k < 16; k++) ef += ea[k] * We[k * HDHID_ + base + c];
    float m = (float)xls[c] + (float)xrd[c] + ef;
    m = m > 0.f ? m : NEG_ * m;
    acc += m * att[base + c];
  }
  logits[t] = acc;
}

// ---------------------------------------------------------------------------
// Per-(node, head) segment softmax over incoming edges (deterministic).
// ---------------------------------------------------------------------------
__global__ void softmax_kernel(const int* __restrict__ rowptr,
                               const int* __restrict__ eids,
                               const int* __restrict__ src,
                               const int* __restrict__ keep,
                               const float* __restrict__ logits,
                               float* __restrict__ alpha) {
  int t = blockIdx.x * blockDim.x + threadIdx.x;
  if (t >= NT_ * HD_) return;
  int n = t >> 2, h = t & 3;
  int b = rowptr[n], e = rowptr[n + 1];
  if (e == b) return;
  float mx = -INFINITY;
  for (int i = b; i < e; i++) mx = fmaxf(mx, logits[eids[i] * HD_ + h]);
  float den = 0.f;
  for (int i = b; i < e; i++) den += expf(logits[eids[i] * HD_ + h] - mx);
  float inv = 1.f / (den + 1e-16f);
  int kn = keep ? keep[n] : 1;
  for (int i = b; i < e; i++) {
    int ed = eids[i];
    int mk = keep ? (kn && keep[src[ed]]) : 1;
    alpha[ed * HD_ + h] = mk ? expf(logits[ed * HD_ + h] - mx) * inv : 0.f;
  }
}

// ---------------------------------------------------------------------------
// Aggregation: block per node; h[n,c] = relu(mean_h(sum_e alpha*xl[src]) + bias)
// ---------------------------------------------------------------------------
__global__ void __launch_bounds__(128)
aggregate_kernel(const _Float16* __restrict__ xl,
                 const float* __restrict__ alpha,
                 const int* __restrict__ rowptr,
                 const int* __restrict__ eids,
                 const int* __restrict__ src,
                 const float* __restrict__ bias,
                 float* __restrict__ hout) {
  __shared__ int   s_src[64];
  __shared__ float s_al[64 * HD_];
  const int n = blockIdx.x;
  const int c = threadIdx.x;
  const int b = rowptr[n], e = rowptr[n + 1];
  float acc[HD_] = {0.f, 0.f, 0.f, 0.f};
  for (int base = b; base < e; base += 64) {
    int cnt = min(64, e - base);
    __syncthreads();
    if (c < cnt) {
      int ed = eids[base + c];
      s_src[c] = src[ed];
#pragma unroll
      for (int h = 0; h < HD_; h++) s_al[c * HD_ + h] = alpha[ed * HD_ + h];
    }
    __syncthreads();
    for (int i = 0; i < cnt; i++) {
      const _Float16* xs = xl + (size_t)s_src[i] * HDHID_ + c;
#pragma unroll
      for (int h = 0; h < HD_; h++)
        acc[h] += s_al[i * HD_ + h] * (float)xs[h * HID_];
    }
  }
  float v = (acc[0] + acc[1] + acc[2] + acc[3]) * 0.25f + bias[c];
  hout[(size_t)n * HID_ + c] = v > 0.f ? v : 0.f;
}

// ---------------------------------------------------------------------------
// TopK pool score: score[n] = h[n]·p / ||p||
// ---------------------------------------------------------------------------
__global__ void score_kernel(const float* __restrict__ h,
                             const float* __restrict__ p,
                             float* __restrict__ scores) {
  int n = blockIdx.x * blockDim.x + threadIdx.x;
  if (n >= NT_) return;
  float nrm = 0.f, s = 0.f;
#pragma unroll 4
  for (int c = 0; c < HID_; c++) { nrm += p[c] * p[c]; s += h[(size_t)n * HID_ + c] * p[c]; }
  scores[n] = s / (sqrtf(nrm) + 1e-16f);
}

// ---------------------------------------------------------------------------
// TopK per graph: exact K-th-largest via 32-step bitwise search on
// sign-flipped float bits; index-order tie-break; gate = tanh(raw score).
// ---------------------------------------------------------------------------
__global__ void __launch_bounds__(1024)
topk_kernel(const float* __restrict__ scores,
            const int* __restrict__ nodeMask, int Kk,
            float* __restrict__ h, int* __restrict__ keepOut) {
  __shared__ unsigned uv[N_];
  __shared__ int cnt;
  __shared__ unsigned char kp[N_];
  const int g = blockIdx.x;
  const int t = threadIdx.x;
  const int gn = g * N_ + t;
  float sraw = scores[gn];
  float smask = (nodeMask && !nodeMask[gn]) ? -INFINITY : sraw;
  unsigned bb = __float_as_uint(smask);
  unsigned u = (bb & 0x80000000u) ? ~bb : (bb | 0x80000000u);
  uv[t] = u;
  __syncthreads();
  unsigned ans = 0;
  for (int bit = 31; bit >= 0; bit--) {
    unsigned trial = ans | (1u << bit);
    if (t == 0) cnt = 0;
    __syncthreads();
    if (u >= trial) atomicAdd(&cnt, 1);
    __syncthreads();
    if (cnt >= Kk) ans = trial;
    __syncthreads();
  }
  if (t == 0) {
    int gt = 0;
    for (int i = 0; i < N_; i++) gt += (uv[i] > ans);
    int need = Kk - gt;
    for (int i = 0; i < N_; i++) {
      if (uv[i] > ans) kp[i] = 1;
      else if (uv[i] == ans && need > 0) { kp[i] = 1; need--; }
      else kp[i] = 0;
    }
  }
  __syncthreads();
  keepOut[gn] = kp[t];
  float gate = tanhf(sraw);
#pragma unroll 4
  for (int c = 0; c < HID_; c++) h[(size_t)gn * HID_ + c] *= gate;
}

// ---------------------------------------------------------------------------
// Global pool + MLP head: one block (128 threads) per graph.
// ---------------------------------------------------------------------------
__global__ void __launch_bounds__(128)
mlp_kernel(const float* __restrict__ h, const int* __restrict__ keep2,
           const float* __restrict__ action,
           const float* __restrict__ Wf1, const float* __restrict__ bf1,
           const float* __restrict__ Wf2, const float* __restrict__ bf2,
           const float* __restrict__ Wf3, const float* __restrict__ bf3,
           float* __restrict__ out) {
  __shared__ float z[2 * HID_ + NR_];
  __shared__ float z1[HID_];
  __shared__ float z2[HID_];
  __shared__ float red[HID_];
  const int b = blockIdx.x;
  const int t = threadIdx.x;
  float mx = -INFINITY, sm = 0.f;
  for (int n = 0; n < N_; n++) {
    int gn = b * N_ + n;
    if (keep2[gn]) {
      float v = h[(size_t)gn * HID_ + t];
      mx = fmaxf(mx, v);
      sm += v;
    }
  }
  z[t] = mx;
  z[HID_ + t] = sm / (float)K2_;
  if (t < NR_) z[2 * HID_ + t] = action[b * NR_ + t];
  __syncthreads();
  float a1 = bf1[t];
  for (int k = 0; k < 2 * HID_ + NR_; k++) a1 += z[k] * Wf1[k * HID_ + t];
  z1[t] = a1 > 0.f ? a1 : 0.f;
  __syncthreads();
  float a2 = bf2[t];
  for (int k = 0; k < HID_; k++) a2 += z1[k] * Wf2[k * HID_ + t];
  z2[t] = a2 > 0.f ? a2 : 0.f;
  __syncthreads();
  red[t] = z2[t] * Wf3[t];
  __syncthreads();
  for (int off = 64; off > 0; off >>= 1) {
    if (t < off) red[t] += red[t + off];
    __syncthreads();
  }
  if (t == 0) out[b] = red[0] + bf3[0];
}

// ---------------------------------------------------------------------------
extern "C" void kernel_launch(void* const* d_in, const int* in_sizes, int n_in,
                              void* d_out, int out_size, void* d_ws, size_t ws_size,
                              hipStream_t stream) {
  (void)in_sizes; (void)n_in; (void)out_size; (void)ws_size;
  const float* x     = (const float*)d_in[0];
  const float* eattr = (const float*)d_in[1];
  const float* action= (const float*)d_in[2];
  const float* W1l = (const float*)d_in[3];  const float* b1l = (const float*)d_in[4];
  const float* W1r = (const float*)d_in[5];  const float* b1r = (const float*)d_in[6];
  const float* W1e = (const float*)d_in[7];  const float* att1= (const float*)d_in[8];
  const float* bias1=(const float*)d_in[9];
  const float* W2l = (const float*)d_in[10]; const float* b2l = (const float*)d_in[11];
  const float* W2r = (const float*)d_in[12]; const float* b2r = (const float*)d_in[13];
  const float* W2e = (const float*)d_in[14]; const float* att2= (const float*)d_in[15];
  const float* bias2=(const float*)d_in[16];
  const float* p1  = (const float*)d_in[17]; const float* p2  = (const float*)d_in[18];
  const float* Wf1 = (const float*)d_in[19]; const float* bf1 = (const float*)d_in[20];
  const float* Wf2 = (const float*)d_in[21]; const float* bf2 = (const float*)d_in[22];
  const float* Wf3 = (const float*)d_in[23]; const float* bf3 = (const float*)d_in[24];
  const int* ei = (const int*)d_in[25];
  const int* src = ei;
  const int* dst = ei + E_;
  float* out = (float*)d_out;

  char* w = (char*)d_ws;
  auto alloc = [&](size_t bytes) -> char* {
    char* p = w; w += (bytes + 255) & ~(size_t)255; return p;
  };
  _Float16* xf16  = (_Float16*)alloc((size_t)NT_ * F_IN_ * 2);
  _Float16* hf16  = (_Float16*)alloc((size_t)NT_ * HID_ * 2);
  _Float16* xlb   = (_Float16*)alloc((size_t)NT_ * HDHID_ * 2);
  _Float16* xrb   = (_Float16*)alloc((size_t)NT_ * HDHID_ * 2);
  _Float16* wpA   = (_Float16*)alloc((size_t)HID_ * HDHID_ * 2);
  _Float16* wpB   = (_Float16*)alloc((size_t)HID_ * HDHID_ * 2);
  float* logits   = (float*)alloc((size_t)E_ * HD_ * 4);
  float* alpha    = (float*)alloc((size_t)E_ * HD_ * 4);
  float* hfeat    = (float*)alloc((size_t)NT_ * HID_ * 4);
  float* scores   = (float*)alloc((size_t)NT_ * 4);
  int* keep1      = (int*)alloc((size_t)NT_ * 4);
  int* keep2      = (int*)alloc((size_t)NT_ * 4);
  int* rowptr     = (int*)alloc((size_t)(NT_ + 1) * 4);
  int* eids       = (int*)alloc((size_t)E_ * 4);

  const int gemmBlocks = (NT_ / 16) * 32 / 8;   // 8192

  // CSR (shared by both layers)
  csr_build_kernel<<<B_, 1024, 0, stream>>>(dst, rowptr, eids);

  // ---------------- Layer 1 ----------------
  cvt_f16_kernel<<<(NT_ * F_IN_) / 256, 256, 0, stream>>>(x, xf16, NT_ * F_IN_);
  pack_w_kernel<<<(F_IN_ * HDHID_ + 255) / 256, 256, 0, stream>>>(W1l, wpA, F_IN_);
  pack_w_kernel<<<(F_IN_ * HDHID_ + 255) / 256, 256, 0, stream>>>(W1r, wpB, F_IN_);
  gemm_wmma_kernel<<<gemmBlocks, 256, 0, stream>>>(xf16, wpA, b1l, xlb, F_IN_);
  gemm_wmma_kernel<<<gemmBlocks, 256, 0, stream>>>(xf16, wpB, b1r, xrb, F_IN_);
  edge_logits_kernel<<<(E_ * HD_) / 256, 256, 0, stream>>>(
      xlb, xrb, eattr, W1e, att1, src, dst, nullptr, logits);
  softmax_kernel<<<(NT_ * HD_) / 256, 256, 0, stream>>>(
      rowptr, eids, src, nullptr, logits, alpha);
  aggregate_kernel<<<NT_, 128, 0, stream>>>(xlb, alpha, rowptr, eids, src, bias1, hfeat);
  score_kernel<<<NT_ / 256, 256, 0, stream>>>(hfeat, p1, scores);
  topk_kernel<<<B_, 1024, 0, stream>>>(scores, nullptr, K1_, hfeat, keep1);

  // ---------------- Layer 2 ----------------
  cvt_f16_kernel<<<(NT_ * HID_) / 256, 256, 0, stream>>>(hfeat, hf16, NT_ * HID_);
  pack_w_kernel<<<(HID_ * HDHID_ + 255) / 256, 256, 0, stream>>>(W2l, wpA, HID_);
  pack_w_kernel<<<(HID_ * HDHID_ + 255) / 256, 256, 0, stream>>>(W2r, wpB, HID_);
  gemm_wmma_kernel<<<gemmBlocks, 256, 0, stream>>>(hf16, wpA, b2l, xlb, HID_);
  gemm_wmma_kernel<<<gemmBlocks, 256, 0, stream>>>(hf16, wpB, b2r, xrb, HID_);
  edge_logits_kernel<<<(E_ * HD_) / 256, 256, 0, stream>>>(
      xlb, xrb, eattr, W2e, att2, src, dst, keep1, logits);
  softmax_kernel<<<(NT_ * HD_) / 256, 256, 0, stream>>>(
      rowptr, eids, src, keep1, logits, alpha);
  aggregate_kernel<<<NT_, 128, 0, stream>>>(xlb, alpha, rowptr, eids, src, bias2, hfeat);
  score_kernel<<<NT_ / 256, 256, 0, stream>>>(hfeat, p2, scores);
  topk_kernel<<<B_, 1024, 0, stream>>>(scores, keep1, K2_, hfeat, keep2);

  // ---------------- Pool + MLP head ----------------
  mlp_kernel<<<B_, 128, 0, stream>>>(hfeat, keep2, action,
                                     Wf1, bf1, Wf2, bf2, Wf3, bf3, out);
}
